// MoEClassifier_74783970558032
// MI455X (gfx1250) — matrix operand order repaired and checked by reading, not compiled, not run on hardware
//
#include <hip/hip_runtime.h>
#include <hip/hip_bf16.h>
#include <math.h>

// MoE classifier for MI455X (gfx1250, wave32, WMMA).
// Pipeline: init -> prep (x->bf16; W1/W2 -> bf16 WMMA-fragment tiles)
//           -> gate (top-3 softmax + token gather)
//           -> expert (async LDS staging + fused W1/GELU/W2 with
//              v_wmma_f32_16x16x32_bf16, fp32 accumulate, software-pipelined
//              B-fragment loads).
// Workspace layout (~137 MB): counts | idx | xb (bf16) | W1 tiles | W2 tiles.

#define B_TOK 16384
#define DDIM  2048
#define HDIM  2048
#define NEXP  6
#define CDIM  1024
#define TOPK  3

#define MT      32            // tokens per expert tile
#define HCHUNK  256           // H processed per chunk
#define XP      (DDIM + 8)    // bf16 LDS pitch for x tile (bank-rotating pad)
#define HP      (HCHUNK + 8)  // bf16 LDS pitch for h chunk

// workspace offsets (bytes)
#define WS_IDX   (1024)
#define WS_XB    (1u << 20)
#define WS_W1T   (68157440u)    // WS_XB + B*D*2
#define WS_W2T   (118489088u)   // WS_W1T + E*H*D*2

typedef __bf16 bf16_t;
typedef __attribute__((ext_vector_type(16))) __bf16 v16bf;
typedef __attribute__((ext_vector_type(8)))  float  v8f;

#define WMMA_BF16(A, B, C) \
  __builtin_amdgcn_wmma_f32_16x16x32_bf16(false, (A), false, (B), (short)0, \
                                          (C), false, false)

// ---------------------------------------------------------------- fragments
// A fragment (16x32 bf16, MxK) from LDS, per CDNA5 ISA layout:
// lane L holds row M=L&15; lanes 0-15: K={kb+0..7, kb+16..23},
// lanes 16-31: K={kb+8..15, kb+24..31}.  Two ds_load_b128 per lane.
__device__ __forceinline__ v16bf load_a_lds(const bf16_t* base, int pitch,
                                            int mbase, int kbase, int lane) {
  const int row  = mbase + (lane & 15);
  const int hsel = (lane >> 4) * 8;
  const bf16_t* p = base + row * pitch + kbase + hsel;
  v16bf a;
  *(uint4*)&a         = *(const uint4*)(p);
  *(((uint4*)&a) + 1) = *(const uint4*)(p + 16);
  return a;
}

// B fragment from pre-tiled bf16 weights: fragment f occupies 1KB,
// lane L's 32 bytes at f*1024 + L*32 (fully coalesced b128 pair).
__device__ __forceinline__ v16bf load_b_tiled(const uint4* __restrict__ Wt,
                                              size_t frag, int lane) {
  const uint4* p = Wt + frag * 64 + lane * 2;
  v16bf b;
  *(uint4*)&b         = p[0];
  *(((uint4*)&b) + 1) = p[1];
  return b;
}

// ---------------------------------------------------------------- init
__global__ __launch_bounds__(256) void init_kernel(float* __restrict__ out,
                                                   int* __restrict__ counts) {
  const size_t stride = (size_t)gridDim.x * blockDim.x;
  const size_t n4 = (size_t)B_TOK * CDIM / 4;
  float4 z; z.x = z.y = z.z = z.w = 0.f;
  for (size_t j = (size_t)blockIdx.x * blockDim.x + threadIdx.x; j < n4; j += stride)
    ((float4*)out)[j] = z;
  if (blockIdx.x == 0 && threadIdx.x < NEXP) counts[threadIdx.x] = 0;
}

// ---------------------------------------------------------------- prep
__global__ __launch_bounds__(256) void prep_x_kernel(const float* __restrict__ x,
                                                     bf16_t* __restrict__ xb) {
  const size_t n = (size_t)B_TOK * DDIM;
  const size_t stride = (size_t)gridDim.x * blockDim.x * 4;
  for (size_t i = ((size_t)blockIdx.x * blockDim.x + threadIdx.x) * 4; i < n;
       i += stride) {
    float4 v = *(const float4*)(x + i);
    bf16_t* d = xb + i;
    d[0] = (bf16_t)v.x; d[1] = (bf16_t)v.y;
    d[2] = (bf16_t)v.z; d[3] = (bf16_t)v.w;
  }
}

// Repack row-major fp32 [R][K] into bf16 WMMA B-fragment tiles.
// frag f = rtile*(K/32)+ktile; lane L holds col n=rtile*16+(L&15),
// k = ktile*32 + (L>=16?8:0) + {0..7, 16..23}; stored at out[f*64 + L*2] (uint4).
__global__ __launch_bounds__(256) void prep_w_kernel(const float* __restrict__ W,
                                                     uint4* __restrict__ out,
                                                     int R, int K) {
  const int wave = threadIdx.x >> 5, lane = threadIdx.x & 31;
  const int ktiles = K >> 5;
  const int nfrag = (R >> 4) * ktiles;
  const int f = blockIdx.x * 8 + wave;
  if (f >= nfrag) return;
  const int kt = f % ktiles, rt = f / ktiles;
  const int n  = rt * 16 + (lane & 15);
  const int kb = kt * 32 + ((lane >> 4) << 3);
  const float* p = W + (size_t)n * K + kb;
  v16bf b;
#pragma unroll
  for (int i = 0; i < 16; ++i) b[i] = (bf16_t)p[i < 8 ? i : i + 8];
  uint4* dst = out + (size_t)f * 64 + lane * 2;
  dst[0] = *(uint4*)&b;
  dst[1] = *(((uint4*)&b) + 1);
}

// ---------------------------------------------------------------- gating
__global__ __launch_bounds__(256) void gate_kernel(
    const float* __restrict__ x, const float* __restrict__ gW,
    const float* __restrict__ gb, float* __restrict__ gw_out,
    int* __restrict__ counts, int* __restrict__ idx) {
  const int wave = threadIdx.x >> 5;
  const int lane = threadIdx.x & 31;
  const int b = blockIdx.x * 8 + wave;
  if (b >= B_TOK) return;

  const float* xr = x + (size_t)b * DDIM;
  float acc[NEXP];
#pragma unroll
  for (int e = 0; e < NEXP; ++e) acc[e] = 0.f;

  for (int k = lane * 4; k < DDIM; k += 32 * 4) {
    float4 xv = *(const float4*)(xr + k);
#pragma unroll
    for (int e = 0; e < NEXP; ++e) {
      float4 wv = *(const float4*)(gW + (size_t)e * DDIM + k);
      acc[e] += xv.x * wv.x + xv.y * wv.y + xv.z * wv.z + xv.w * wv.w;
    }
  }
  float logit[NEXP];
#pragma unroll
  for (int e = 0; e < NEXP; ++e) {
    float v = acc[e];
#pragma unroll
    for (int off = 16; off > 0; off >>= 1) v += __shfl_xor(v, off, 32);
    logit[e] = v + gb[e];
  }

  // top-3 (first index wins ties, matching lax.top_k stability)
  int sel[TOPK]; float sv[TOPK]; bool used[NEXP];
#pragma unroll
  for (int e = 0; e < NEXP; ++e) used[e] = false;
#pragma unroll
  for (int k = 0; k < TOPK; ++k) {
    int best = 0; float bv = -3.402823466e38f;
#pragma unroll
    for (int e = 0; e < NEXP; ++e)
      if (!used[e] && logit[e] > bv) { bv = logit[e]; best = e; }
    used[best] = true; sel[k] = best; sv[k] = bv;
  }
  const float m = sv[0];
  float ex[TOPK], s = 0.f;
#pragma unroll
  for (int k = 0; k < TOPK; ++k) { ex[k] = expf(sv[k] - m); s += ex[k]; }
  const float inv = 1.f / s;
  float g[NEXP];
#pragma unroll
  for (int e = 0; e < NEXP; ++e) g[e] = 0.f;
#pragma unroll
  for (int k = 0; k < TOPK; ++k) g[sel[k]] = ex[k] * inv;

  if (lane == 0) {
#pragma unroll
    for (int e = 0; e < NEXP; ++e) gw_out[(size_t)b * NEXP + e] = g[e];
#pragma unroll
    for (int k = 0; k < TOPK; ++k) {
      const int e = sel[k];
      const int pos = atomicAdd(&counts[e], 1);
      idx[e * B_TOK + pos] = b;
    }
  }
}

// ---------------------------------------------------------------- experts
__global__ __launch_bounds__(256) void expert_kernel(
    const bf16_t* __restrict__ xb, const uint4* __restrict__ W1t,
    const float* __restrict__ b1, const uint4* __restrict__ W2t,
    const float* __restrict__ b2, const float* __restrict__ gw,
    const int* __restrict__ counts, const int* __restrict__ idx,
    float* __restrict__ out) {
  extern __shared__ char smem[];
  bf16_t* xs = (bf16_t*)smem;                  // [MT][XP]
  bf16_t* hs = xs + (size_t)MT * XP;           // [MT][HP]
  int*   tok = (int*)(hs + (size_t)MT * HP);   // [MT]
  float* wgt = (float*)(tok + MT);             // [MT]

  const int e   = blockIdx.y;
  const int n_e = counts[e];
  const int base = blockIdx.x * MT;
  if (base >= n_e) return;

  const int wave = threadIdx.x >> 5;
  const int lane = threadIdx.x & 31;

  if (threadIdx.x < MT) {
    const int r = base + threadIdx.x;
    const int t = (r < n_e) ? idx[e * B_TOK + r] : -1;
    tok[threadIdx.x] = t;
    wgt[threadIdx.x] = (t >= 0) ? gw[(size_t)t * NEXP + e] : 0.f;
  }
  __syncthreads();

  // stage x tile: async global(bf16) -> LDS, 4 rows per wave, 16B per lane
  for (int r = wave; r < MT; r += 8) {
    const int t = tok[r];
    bf16_t* dst = xs + (size_t)r * XP;
    if (t >= 0) {
      const bf16_t* src = xb + (size_t)t * DDIM;
#pragma unroll
      for (int k = 0; k < DDIM; k += 256) {  // 32 lanes x 8 halfs
        const uint32_t loff =
            (uint32_t)((const char*)(dst + k + lane * 8) - (const char*)smem);
        const void* g = (const void*)(src + k + lane * 8);
        asm volatile("global_load_async_to_lds_b128 %0, %1, off"
                     :: "v"(loff), "v"(g) : "memory");
      }
    } else {
      uint4 z; z.x = z.y = z.z = z.w = 0u;
      for (int k = lane * 8; k < DDIM; k += 256) *(uint4*)(dst + k) = z;
    }
  }
  asm volatile("s_wait_asynccnt 0" ::: "memory");
  __syncthreads();

  v8f zero = {};
  v8f oacc[2][8];
#pragma unroll
  for (int mt = 0; mt < 2; ++mt)
#pragma unroll
    for (int nt = 0; nt < 8; ++nt) oacc[mt][nt] = zero;

  const int ncol   = lane & 15;
  const int rowoff = (lane < 16) ? 0 : 8;

  for (int hc = 0; hc < HDIM / HCHUNK; ++hc) {
    // ---- GEMM1: this wave computes h[:, hc*256 + wave*32 .. +31] ----
    const int nb = wave * 32;  // column base within chunk
    // W1 fragment row-tile: e*(H/16) + (hc*256 + nb + nt*16)/16
    const size_t rt1 = (size_t)e * (HDIM / 16) + hc * (HCHUNK / 16) + wave * 2;
    const size_t fb0 = (rt1 + 0) * (DDIM / 32);
    const size_t fb1 = (rt1 + 1) * (DDIM / 32);
    v8f hacc[2][2];
#pragma unroll
    for (int mt = 0; mt < 2; ++mt)
#pragma unroll
      for (int nt = 0; nt < 2; ++nt) hacc[mt][nt] = zero;

    // software-pipelined B fragments (double buffer per n-tile)
    v16bf bc0 = load_b_tiled(W1t, fb0, lane);
    v16bf bc1 = load_b_tiled(W1t, fb1, lane);
#pragma unroll 2
    for (int kk = 0; kk < DDIM; kk += 32) {
      v16bf a0 = load_a_lds(xs, XP, 0, kk, lane);
      v16bf a1 = load_a_lds(xs, XP, 16, kk, lane);
      const int ktn = (kk + 32 < DDIM) ? ((kk >> 5) + 1) : (kk >> 5);
      v16bf bn0 = load_b_tiled(W1t, fb0 + ktn, lane);
      v16bf bn1 = load_b_tiled(W1t, fb1 + ktn, lane);
      hacc[0][0] = WMMA_BF16(a0, bc0, hacc[0][0]);
      hacc[1][0] = WMMA_BF16(a1, bc0, hacc[1][0]);
      hacc[0][1] = WMMA_BF16(a0, bc1, hacc[0][1]);
      hacc[1][1] = WMMA_BF16(a1, bc1, hacc[1][1]);
      bc0 = bn0; bc1 = bn1;
    }
    // bias + exact GELU -> hs (bf16)
#pragma unroll
    for (int nt = 0; nt < 2; ++nt) {
      const int col = nb + nt * 16 + ncol;  // within chunk
      const float bias = b1[(size_t)e * HDIM + hc * HCHUNK + col];
#pragma unroll
      for (int mt = 0; mt < 2; ++mt) {
#pragma unroll
        for (int i = 0; i < 8; ++i) {
          const float v = hacc[mt][nt][i] + bias;
          const float gv = 0.5f * v * (1.0f + erff(v * 0.70710678118654752f));
          hs[(size_t)(mt * 16 + rowoff + i) * HP + col] = (bf16_t)gv;
        }
      }
    }

    // ---- GEMM2: this wave accumulates out[:, wave*128 .. +127] ----
    // W2 fragment row-tile: e*(C/16) + wave*8 + nt; 2-deep fragment pipeline
    // over the flattened 64-step (k x nt) sequence; first two issued before
    // the barrier (independent of hs).
    const size_t rt2 = (size_t)e * (CDIM / 16) + wave * 8;
    const size_t ktb = (size_t)hc * (HCHUNK / 32);
#define F2(s) ((rt2 + ((s) & 7)) * (HDIM / 32) + ktb + ((s) >> 3))
    v16bf bcur = load_b_tiled(W2t, F2(0), lane);
    v16bf bnxt = load_b_tiled(W2t, F2(1), lane);
    __syncthreads();  // hs now visible

#pragma unroll
    for (int kk = 0; kk < HCHUNK; kk += 32) {
      v16bf a0 = load_a_lds(hs, HP, 0, kk, lane);
      v16bf a1 = load_a_lds(hs, HP, 16, kk, lane);
#pragma unroll
      for (int nt = 0; nt < 8; ++nt) {
        const int s = (kk >> 5) * 8 + nt;
        v16bf bpre;
        if (s + 2 < 64) bpre = load_b_tiled(W2t, F2(s + 2), lane);
        oacc[0][nt] = WMMA_BF16(a0, bcur, oacc[0][nt]);
        oacc[1][nt] = WMMA_BF16(a1, bcur, oacc[1][nt]);
        bcur = bnxt;
        if (s + 2 < 64) bnxt = bpre;
      }
    }
#undef F2
    __syncthreads();  // hs is rewritten next chunk
  }

  // ---- epilogue: weighted scatter-add into out ----
#pragma unroll
  for (int nt = 0; nt < 8; ++nt) {
    const int c = wave * 128 + nt * 16 + ncol;
    const float bias2 = b2[(size_t)e * CDIM + c];
#pragma unroll
    for (int mt = 0; mt < 2; ++mt) {
#pragma unroll
      for (int i = 0; i < 8; ++i) {
        const int r = mt * 16 + rowoff + i;
        const int t = tok[r];
        if (t >= 0) {
          atomicAdd(out + (size_t)t * CDIM + c,
                    wgt[r] * (oacc[mt][nt][i] + bias2));
        }
      }
    }
  }
}

// ---------------------------------------------------------------- launch
extern "C" void kernel_launch(void* const* d_in, const int* in_sizes, int n_in,
                              void* d_out, int out_size, void* d_ws,
                              size_t ws_size, hipStream_t stream) {
  const float* x  = (const float*)d_in[0];
  const float* gW = (const float*)d_in[1];
  const float* gb = (const float*)d_in[2];
  const float* W1 = (const float*)d_in[3];
  const float* b1 = (const float*)d_in[4];
  const float* W2 = (const float*)d_in[5];
  const float* b2 = (const float*)d_in[6];

  float* out    = (float*)d_out;
  float* gw_out = out + (size_t)B_TOK * CDIM;  // second tuple output [B, E]

  int*    counts = (int*)d_ws;
  int*    idx    = (int*)((char*)d_ws + WS_IDX);
  bf16_t* xb     = (bf16_t*)((char*)d_ws + WS_XB);
  uint4*  W1t    = (uint4*)((char*)d_ws + WS_W1T);
  uint4*  W2t    = (uint4*)((char*)d_ws + WS_W2T);

  init_kernel<<<2048, 256, 0, stream>>>(out, counts);
  prep_x_kernel<<<4096, 256, 0, stream>>>(x, xb);
  // W1: 6*2048 rows x 2048 cols -> 49152 fragments; W2: 6*1024 rows -> 24576
  prep_w_kernel<<<(NEXP * HDIM / 16) * (DDIM / 32) / 8, 256, 0, stream>>>(
      W1, W1t, NEXP * HDIM, DDIM);
  prep_w_kernel<<<(NEXP * CDIM / 16) * (HDIM / 32) / 8, 256, 0, stream>>>(
      W2, W2t, NEXP * CDIM, HDIM);
  gate_kernel<<<B_TOK / 8, 256, 0, stream>>>(x, gW, gb, gw_out, counts, idx);

  const size_t smem = (size_t)MT * XP * sizeof(bf16_t) +
                      (size_t)MT * HP * sizeof(bf16_t) +
                      MT * sizeof(int) + MT * sizeof(float);
  dim3 grid(B_TOK / MT, NEXP);
  expert_kernel<<<grid, 256, smem, stream>>>(xb, W1t, b1, W2t, b2, gw_out,
                                             counts, idx, out);
}
